// TransitionModel_18287970747025
// MI455X (gfx1250) — compile-verified, matrix operand order checked by reading, not run on hardware
//
#include <hip/hip_runtime.h>
#include <hip/hip_bf16.h>

// ---------------------------------------------------------------------------
// RSSM transition rollout for MI455X (gfx1250), bf16 WMMA + fp32 accumulate.
//
// Roofline: ~330 GFLOP total, weights ~20MB bf16 (fully L2-resident on the
// 192MB L2), HBM traffic ~180MB -> neither HBM- nor FLOP-bound; the critical
// resource is L2->WGP fragment traffic on the serialized per-step GEMMs.
// Strategy: v_wmma_f32_16x16x32_bf16 everywhere; shared weight (B) panels are
// staged in LDS once per 8-wave workgroup (double-buffered, one barrier per
// K-step) cutting B-fragment amplification 16x -> 2x; activation (A)
// fragments load straight from global/L2 (two b128 per lane, per the CDNA5
// 16-bit A-fragment layout); heavy epilogue fusion (GRU gates, softplus
// sampling) keeps kernel count at 7 per step.
// ---------------------------------------------------------------------------

typedef __attribute__((ext_vector_type(16))) __bf16 v16bf;
typedef __attribute__((ext_vector_type(8)))  __bf16 v8bf;
typedef __attribute__((ext_vector_type(8)))  float  v8f;

#define BB   256   // batch
#define SDIM 128   // state size
#define ADIM 32    // action size
#define BEL  1024  // belief size
#define HID  1024
#define EMB  1024
#define TT   64    // T-1 steps

__device__ __forceinline__ v8f wmma_bf16(v16bf a, v16bf b, v8f c) {
  // (neg_a, A, neg_b, B, c_mod, C, reuse_a, reuse_b)
  return __builtin_amdgcn_wmma_f32_16x16x32_bf16(false, a, false, b, (short)0, c,
                                                 false, false);
}

// 16x32 bf16 fragment from a row-major matrix in global memory (stride ld).
// Lanes 0-15 hold row r0+lane, K {k0..k0+7, k0+16..k0+23}; lanes 16-31 hold
// the same rows, K {k0+8..15, k0+24..31}.
__device__ __forceinline__ v16bf load_frag(const __bf16* __restrict__ base,
                                           int r0, int ld, int k0) {
  int lane = threadIdx.x & 31;
  int r    = r0 + (lane & 15);
  int kk   = k0 + ((lane & 16) ? 8 : 0);
  const __bf16* p = base + (size_t)r * ld + kk;
  v8bf lo = *(const v8bf*)(p);        // global_load_b128
  v8bf hi = *(const v8bf*)(p + 16);   // global_load_b128
  v16bf out;
#pragma unroll
  for (int i = 0; i < 8; ++i) { out[i] = lo[i]; out[i + 8] = hi[i]; }
  return out;
}

// Same fragment pattern, but from an LDS panel laid out [rows][32] bf16.
__device__ __forceinline__ v16bf lds_frag(const __bf16* buf, int r0) {
  int lane = threadIdx.x & 31;
  const __bf16* p = buf + (r0 + (lane & 15)) * 32 + ((lane & 16) ? 8 : 0);
  v8bf lo = *(const v8bf*)(p);        // ds_load_b128
  v8bf hi = *(const v8bf*)(p + 16);   // ds_load_b128
  v16bf out;
#pragma unroll
  for (int i = 0; i < 8; ++i) { out[i] = lo[i]; out[i + 8] = hi[i]; }
  return out;
}

__device__ __forceinline__ float sigmoidf_(float x) { return 1.f / (1.f + __expf(-x)); }
__device__ __forceinline__ float softplusf_(float x) {
  return (x > 20.f) ? x : log1pf(__expf(x));
}

// ---------------------------------------------------------------------------
// fp32 -> bf16 weight conversion (once per call)
// ---------------------------------------------------------------------------
__global__ __launch_bounds__(256) void cvt_kernel(const float* __restrict__ src,
                                                  __bf16* __restrict__ dst, int n) {
  int i = blockIdx.x * 256 + threadIdx.x;
  if (i < n) dst[i] = (__bf16)src[i];
}

// ---------------------------------------------------------------------------
// Per-step input prep: masked state/belief, action concat, obs bf16 convert.
// ---------------------------------------------------------------------------
__global__ __launch_bounds__(256) void mask_kernel(
    const float* __restrict__ state_src,   // [BB,SDIM]
    const float* __restrict__ belief_src,  // [BB,BEL]
    const float* __restrict__ act,         // [BB,ADIM]
    const float* __restrict__ obs,         // [BB,EMB]
    const float* __restrict__ nt_src,      // [BB] or nullptr (==1.0)
    __bf16* __restrict__ xin,              // [BB,160]  bf16 [s*nt, a]
    __bf16* __restrict__ hb,               // [BB,BEL]  bf16 belief*nt
    float*  __restrict__ hf,               // [BB,BEL]  f32  belief*nt
    __bf16* __restrict__ nbobs)            // [BB,2048] cols 1024.. = obs bf16
{
  int tid = blockIdx.x * 256 + threadIdx.x;
  int b = tid >> 10, j = tid & 1023;
  float nt = nt_src ? nt_src[b] : 1.f;
  float h = belief_src[(size_t)b * BEL + j] * nt;
  hf[(size_t)b * BEL + j] = h;
  hb[(size_t)b * BEL + j] = (__bf16)h;
  nbobs[(size_t)b * 2048 + 1024 + j] = (__bf16)obs[(size_t)b * EMB + j];
  if (j < SDIM)
    xin[(size_t)b * 160 + j] = (__bf16)(state_src[(size_t)b * SDIM + j] * nt);
  else if (j < SDIM + ADIM)
    xin[(size_t)b * 160 + j] = (__bf16)act[(size_t)b * ADIM + (j - SDIM)];
}

// ---------------------------------------------------------------------------
// Out = relu(A @ W^T + bias) -> bf16.  W stored [N,K] row-major.
// Block = 8 waves: one 64-col N-panel shared via LDS, 128 M-rows (16/wave).
// B panel (64 x 32 bf16 = 4KB) double-buffered in LDS; one barrier / K-step.
// ---------------------------------------------------------------------------
__device__ __forceinline__ void store_relu_frag(v8f acc, const float* __restrict__ bias,
                                                __bf16* __restrict__ Out, int ldo,
                                                int m0, int n0, int col, int rb) {
#pragma unroll
  for (int i = 0; i < 8; ++i) {
    int row = m0 + rb + i;
    int c   = n0 + col;
    float v = fmaxf(acc[i] + bias[c], 0.f);
    Out[(size_t)row * ldo + c] = (__bf16)v;
  }
}

__global__ __launch_bounds__(256) void gemm_relu_kernel(
    const __bf16* __restrict__ A, int lda,       // [M,K] (lda >= K)
    const __bf16* __restrict__ W,                // [N,K] row-major
    const float*  __restrict__ bias,             // [N]
    __bf16* __restrict__ Out, int ldo,           // [M,N]
    int M, int N, int K)
{
  __shared__ __bf16 lbuf[2][64 * 32];            // 2 x 4KB B panel
  int t    = threadIdx.x;
  int wave = t >> 5;
  int lane = t & 31;
  int nblks = N >> 6;
  int mblk  = blockIdx.x / nblks;
  int n0    = (blockIdx.x % nblks) << 6;
  int m0    = (mblk << 7) + (wave << 4);
  (void)M;

  // staging: 64 rows x 4 chunks(16B) = 256 chunks, one per thread
  int srow = t >> 2, sch = (t & 3) << 3;         // 8 bf16 per chunk
  const __bf16* gsrc = W + (size_t)(n0 + srow) * K + sch;
  __bf16* ldst0 = &lbuf[0][srow * 32 + sch];
  __bf16* ldst1 = &lbuf[1][srow * 32 + sch];

  *(v8bf*)ldst0 = *(const v8bf*)(gsrc);          // prologue stage 0
  __syncthreads();

  v8f a0 = {}, a1 = {}, a2 = {}, a3 = {};
  int nsteps = K >> 5;
  for (int s = 0; s < nsteps; ++s) {
    int k0 = s << 5;
    v8bf nxt = {};
    bool more = (s + 1 < nsteps);
    if (more) nxt = *(const v8bf*)(gsrc + (s + 1) * 32);  // overlap with WMMA
    const __bf16* cur = lbuf[s & 1];
    v16bf a  = load_frag(A, m0, lda, k0);
    v16bf b0 = lds_frag(cur,  0);
    v16bf b1 = lds_frag(cur, 16);
    v16bf b2 = lds_frag(cur, 32);
    v16bf b3 = lds_frag(cur, 48);
    a0 = wmma_bf16(a, b0, a0);
    a1 = wmma_bf16(a, b1, a1);
    a2 = wmma_bf16(a, b2, a2);
    a3 = wmma_bf16(a, b3, a3);
    if (more) *(v8bf*)((s & 1) ? ldst0 : ldst1) = nxt;
    __syncthreads();
  }
  int col = lane & 15;
  int rb  = (lane & 16) ? 8 : 0;
  store_relu_frag(a0, bias, Out, ldo, m0, n0 +  0, col, rb);
  store_relu_frag(a1, bias, Out, ldo, m0, n0 + 16, col, rb);
  store_relu_frag(a2, bias, Out, ldo, m0, n0 + 32, col, rb);
  store_relu_frag(a3, bias, Out, ldo, m0, n0 + 48, col, rb);
}

// ---------------------------------------------------------------------------
// Fused GRU cell. Block = 8 waves (128 M-rows) sharing one 16-col J-tile.
// LDS stages the six 16-row gate panels (ir,iz,in from Wih; hr,hz,hn from
// Whh) = 96 rows x 32 k = 6KB, double-buffered. Epilogue applies the full
// gate math + belief blend in-register.
// ---------------------------------------------------------------------------
__global__ __launch_bounds__(256) void gru_kernel(
    const __bf16* __restrict__ X,    // [BB,BEL]  relu(Wsa [s,a])
    const __bf16* __restrict__ Hb,   // [BB,BEL]  masked belief bf16
    const float*  __restrict__ Hf,   // [BB,BEL]  masked belief f32
    const __bf16* __restrict__ Wih,  // [3*BEL,BEL]
    const __bf16* __restrict__ Whh,  // [3*BEL,BEL]
    const float*  __restrict__ bih,
    const float*  __restrict__ bhh,
    float*  __restrict__ belief_out, // d_out beliefs[t]  [BB,BEL]
    __bf16* __restrict__ nbobs)      // [BB,2048] cols 0..1023 <- nb bf16
{
  __shared__ __bf16 lbuf[2][96 * 32];            // 2 x 6KB gate panels
  int t = threadIdx.x, wave = t >> 5, lane = t & 31;
  int mblk = blockIdx.x >> 6;                    // 2 M-blocks
  int j0   = (blockIdx.x & 63) << 4;             // 64 J-tiles
  int m0   = (mblk << 7) + (wave << 4);

  // staging: 96 rows x 4 chunks = 384 chunks; thread t does chunk t (+t+256)
  auto chunk_src = [&](int c) -> const __bf16* {
    int lrow = c >> 2;                 // 0..95
    int ch   = (c & 3) << 3;
    int p    = lrow >> 4;              // 0..5 : ir,iz,in,hr,hz,hn
    int r    = lrow & 15;
    const __bf16* mat = (p < 3) ? Wih : Whh;
    int po = (p < 3) ? p : p - 3;
    return mat + (size_t)(po * BEL + j0 + r) * BEL + ch;
  };
  bool do2 = (t < 128);
  int  c2  = do2 ? (t + 256) : 383;
  const __bf16* src1 = chunk_src(t);
  const __bf16* src2 = chunk_src(c2);
  int l1 = (t >> 2) * 32 + ((t & 3) << 3);
  int l2 = (c2 >> 2) * 32 + ((c2 & 3) << 3);

  *(v8bf*)&lbuf[0][l1] = *(const v8bf*)src1;
  if (do2) *(v8bf*)&lbuf[0][l2] = *(const v8bf*)src2;
  __syncthreads();                                // exec reconverged here

  v8f ir = {}, iz = {}, in_ = {}, hr = {}, hz = {}, hn = {};
  for (int s = 0; s < 32; ++s) {
    int k0 = s << 5;
    v8bf n1 = {}, n2 = {};
    bool more = (s + 1 < 32);
    if (more) {
      n1 = *(const v8bf*)(src1 + (s + 1) * 32);
      if (do2) n2 = *(const v8bf*)(src2 + (s + 1) * 32);
    }
    const __bf16* cur = lbuf[s & 1];
    v16bf ax = load_frag(X,  m0, BEL, k0);
    v16bf ah = load_frag(Hb, m0, BEL, k0);
    ir  = wmma_bf16(ax, lds_frag(cur,  0), ir);
    iz  = wmma_bf16(ax, lds_frag(cur, 16), iz);
    in_ = wmma_bf16(ax, lds_frag(cur, 32), in_);
    hr  = wmma_bf16(ah, lds_frag(cur, 48), hr);
    hz  = wmma_bf16(ah, lds_frag(cur, 64), hz);
    hn  = wmma_bf16(ah, lds_frag(cur, 80), hn);
    if (more) {
      __bf16* dst = lbuf[(s + 1) & 1];
      *(v8bf*)&dst[l1] = n1;
      if (do2) *(v8bf*)&dst[l2] = n2;
    }
    __syncthreads();
  }
  int col = lane & 15;
  int rb  = (lane & 16) ? 8 : 0;
#pragma unroll
  for (int i = 0; i < 8; ++i) {
    int row = m0 + rb + i;
    int j   = j0 + col;
    float r = sigmoidf_(ir[i] + bih[j] + hr[i] + bhh[j]);
    float z = sigmoidf_(iz[i] + bih[BEL + j] + hz[i] + bhh[BEL + j]);
    float n = tanhf(in_[i] + bih[2 * BEL + j] + r * (hn[i] + bhh[2 * BEL + j]));
    float h = Hf[(size_t)row * BEL + j];
    float nb = (1.f - z) * n + z * h;
    belief_out[(size_t)row * BEL + j] = nb;
    nbobs[(size_t)row * 2048 + j] = (__bf16)nb;
  }
}

// ---------------------------------------------------------------------------
// Distribution head: [mean | std_raw] = H @ Wh^T + b; std = softplus+0.1;
// state = mean + std*noise. Small (0.5MB weights) -> direct global fragments.
// ---------------------------------------------------------------------------
__global__ __launch_bounds__(256) void head_kernel(
    const __bf16* __restrict__ H,      // [BB,HID]
    const __bf16* __restrict__ Wh,     // [2*SDIM,HID]
    const float*  __restrict__ bh,     // [2*SDIM]
    const float*  __restrict__ noise,  // [BB,SDIM]
    float* __restrict__ means, float* __restrict__ stds, float* __restrict__ states)
{
  int wave = (blockIdx.x << 3) + (threadIdx.x >> 5);
  int lane = threadIdx.x & 31;
  int m0 = (wave >> 3) << 4;   // 16 M-tiles
  int j0 = (wave & 7) << 4;    // 8 J-tiles (SDIM=128)
  v8f am = {}, as = {};
  for (int k0 = 0; k0 < HID; k0 += 32) {
    v16bf a = load_frag(H, m0, HID, k0);
    am = wmma_bf16(a, load_frag(Wh,        j0, HID, k0), am);
    as = wmma_bf16(a, load_frag(Wh, SDIM + j0, HID, k0), as);
  }
  int col = lane & 15;
  int rb  = (lane & 16) ? 8 : 0;
#pragma unroll
  for (int i = 0; i < 8; ++i) {
    int row = m0 + rb + i;
    int j   = j0 + col;
    float m  = am[i] + bh[j];
    float sd = softplusf_(as[i] + bh[SDIM + j]) + 0.1f;
    float st = m + sd * noise[(size_t)row * SDIM + j];
    means [(size_t)row * SDIM + j] = m;
    stds  [(size_t)row * SDIM + j] = sd;
    states[(size_t)row * SDIM + j] = st;
  }
}

// ---------------------------------------------------------------------------
// Host driver
// ---------------------------------------------------------------------------
extern "C" void kernel_launch(void* const* d_in, const int* in_sizes, int n_in,
                              void* d_out, int out_size, void* d_ws, size_t ws_size,
                              hipStream_t stream) {
  (void)in_sizes; (void)n_in; (void)out_size; (void)ws_size;
  const float* prev_state   = (const float*)d_in[0];
  const float* actions      = (const float*)d_in[1];
  const float* prev_belief  = (const float*)d_in[2];
  const float* observations = (const float*)d_in[3];
  const float* nonterm      = (const float*)d_in[4];
  const float* prior_noise  = (const float*)d_in[5];
  const float* post_noise   = (const float*)d_in[6];
  const float* W_sa = (const float*)d_in[7];  const float* b_sa = (const float*)d_in[8];
  const float* W_ih = (const float*)d_in[9];  const float* W_hh = (const float*)d_in[10];
  const float* b_ih = (const float*)d_in[11]; const float* b_hh = (const float*)d_in[12];
  const float* W_bp = (const float*)d_in[13]; const float* b_bp = (const float*)d_in[14];
  const float* W_sp = (const float*)d_in[15]; const float* b_sp = (const float*)d_in[16];
  const float* W_bq = (const float*)d_in[17]; const float* b_bq = (const float*)d_in[18];
  const float* W_sq = (const float*)d_in[19]; const float* b_sq = (const float*)d_in[20];

  // Output slices (return order): beliefs, prior_states, prior_means,
  // prior_stds, post_states, post_means, post_stds.
  float* out = (float*)d_out;
  const size_t nBel = (size_t)TT * BB * BEL;
  const size_t nS   = (size_t)TT * BB * SDIM;
  float* beliefs      = out;
  float* prior_states = out + nBel;
  float* prior_means  = prior_states + nS;
  float* prior_stds   = prior_means + nS;
  float* post_states  = prior_stds + nS;
  float* post_means   = post_states + nS;
  float* post_stds    = post_means + nS;

  // Workspace carve (~25 MB).
  size_t off = 0;
  auto carve = [&](size_t bytes) -> void* {
    void* r = (char*)d_ws + off;
    off = (off + bytes + 255) & ~(size_t)255;
    return r;
  };
  __bf16* Wsa_b = (__bf16*)carve((size_t)BEL * 160 * 2);
  __bf16* Wih_b = (__bf16*)carve((size_t)3 * BEL * BEL * 2);
  __bf16* Whh_b = (__bf16*)carve((size_t)3 * BEL * BEL * 2);
  __bf16* Wbp_b = (__bf16*)carve((size_t)HID * BEL * 2);
  __bf16* Wsp_b = (__bf16*)carve((size_t)2 * SDIM * HID * 2);
  __bf16* Wbq_b = (__bf16*)carve((size_t)HID * (BEL + EMB) * 2);
  __bf16* Wsq_b = (__bf16*)carve((size_t)2 * SDIM * HID * 2);
  __bf16* xin   = (__bf16*)carve((size_t)BB * 160 * 2);
  __bf16* xbf   = (__bf16*)carve((size_t)BB * BEL * 2);
  __bf16* hb    = (__bf16*)carve((size_t)BB * BEL * 2);
  float*  hf    = (float*) carve((size_t)BB * BEL * 4);
  __bf16* nbobs = (__bf16*)carve((size_t)BB * 2048 * 2);  // [nb | obs] bf16
  __bf16* hp    = (__bf16*)carve((size_t)BB * HID * 2);
  __bf16* hq    = (__bf16*)carve((size_t)BB * HID * 2);

  auto cvt = [&](const float* s, __bf16* d, int n) {
    cvt_kernel<<<(n + 255) / 256, 256, 0, stream>>>(s, d, n);
  };
  cvt(W_sa, Wsa_b, BEL * 160);
  cvt(W_ih, Wih_b, 3 * BEL * BEL);
  cvt(W_hh, Whh_b, 3 * BEL * BEL);
  cvt(W_bp, Wbp_b, HID * BEL);
  cvt(W_sp, Wsp_b, 2 * SDIM * HID);
  cvt(W_bq, Wbq_b, HID * (BEL + EMB));
  cvt(W_sq, Wsq_b, 2 * SDIM * HID);

  for (int t = 0; t < TT; ++t) {
    // carries read from the d_out slices written in step t-1 (deterministic)
    const float* s_src  = (t == 0) ? prev_state  : post_states + (size_t)(t - 1) * BB * SDIM;
    const float* be_src = (t == 0) ? prev_belief : beliefs     + (size_t)(t - 1) * BB * BEL;
    const float* nt_src = (t == 0) ? nullptr     : nonterm     + (size_t)(t - 1) * BB;

    mask_kernel<<<(BB * 1024) / 256, 256, 0, stream>>>(
        s_src, be_src, actions + (size_t)t * BB * ADIM,
        observations + (size_t)t * BB * EMB, nt_src, xin, hb, hf, nbobs);

    // x = relu([s,a] @ W_sa^T + b_sa)          M=256 N=1024 K=160
    gemm_relu_kernel<<<32, 256, 0, stream>>>(xin, 160, Wsa_b, b_sa, xbf, BEL,
                                             BB, BEL, 160);
    // GRU: nb = GRU(x, h)
    gru_kernel<<<128, 256, 0, stream>>>(xbf, hb, hf, Wih_b, Whh_b, b_ih, b_hh,
                                        beliefs + (size_t)t * BB * BEL, nbobs);
    // prior: hp = relu(nb @ W_bp^T + b_bp)     M=256 N=1024 K=1024
    gemm_relu_kernel<<<32, 256, 0, stream>>>(nbobs, 2048, Wbp_b, b_bp, hp, HID,
                                             BB, HID, BEL);
    head_kernel<<<16, 256, 0, stream>>>(hp, Wsp_b, b_sp,
                                        prior_noise + (size_t)t * BB * SDIM,
                                        prior_means + (size_t)t * BB * SDIM,
                                        prior_stds  + (size_t)t * BB * SDIM,
                                        prior_states + (size_t)t * BB * SDIM);
    // posterior: hq = relu([nb,obs] @ W_bq^T)  M=256 N=1024 K=2048
    gemm_relu_kernel<<<32, 256, 0, stream>>>(nbobs, 2048, Wbq_b, b_bq, hq, HID,
                                             BB, HID, BEL + EMB);
    head_kernel<<<16, 256, 0, stream>>>(hq, Wsq_b, b_sq,
                                        post_noise + (size_t)t * BB * SDIM,
                                        post_means + (size_t)t * BB * SDIM,
                                        post_stds  + (size_t)t * BB * SDIM,
                                        post_states + (size_t)t * BB * SDIM);
  }
}
